// TacoSemiProtoNet_14791867367795
// MI455X (gfx1250) — matrix operand order repaired and checked by reading, not compiled, not run on hardware
//
#include <hip/hip_runtime.h>
#include <hip/hip_bf16.h>

typedef __attribute__((ext_vector_type(16))) _Float16 v16h;
typedef __attribute__((ext_vector_type(8)))  _Float16 v8h;
typedef __attribute__((ext_vector_type(8)))  float    v8f;

// ---------------------------------------------------------------------------
// Convert OIHW fp32 weights to f16 with BN scale folded in, K-padded.
// w16[ch][k] = w[ch][k] * scale[ch]  (k < K), else 0.
// ---------------------------------------------------------------------------
__global__ void k_cvt_weights(const float* __restrict__ w, const float* __restrict__ scale,
                              _Float16* __restrict__ w16, int K, int Kpad) {
  int i = blockIdx.x * blockDim.x + threadIdx.x;
  if (i >= 64 * Kpad) return;
  int ch = i / Kpad, k = i - ch * Kpad;
  w16[i] = (k < K) ? (_Float16)(w[ch * K + k] * scale[ch]) : (_Float16)0.0f;
}

// ---------------------------------------------------------------------------
// Fused conv3x3(SAME) + BN bias + ReLU + maxpool2x2, WMMA implicit GEMM.
// block = 128 threads (4 waves); one image, 4x4 output tile, 64 out channels.
// wave w -> channels [16w, 16w+16).
//
// Per CDNA5 f16 fragment layouts (ISA 7.12.2), each lane's K values are
// contiguous:  A: k = kc*32 + hl*8 + e (e<8), +16 for e>=8  -> two b128 loads
//              B: k = kc*32 + hl*8 + e (e<8), +16 for e>=8  -> two 16B LDS reads
// so we build the im2col matrix B_lds[n][k] once (decode paid once, no
// divergence in the hot loop) and the K-loop is pure vector loads + wmma.
// ---------------------------------------------------------------------------
template <int CIN, int KPAD, bool LAYER1>
__global__ __launch_bounds__(128)
void k_conv_bn_relu_pool(const float* __restrict__ shot,
                         const float* __restrict__ query,
                         const float* __restrict__ unlabel,
                         const _Float16* __restrict__ actIn,
                         const _Float16* __restrict__ w16,
                         const float* __restrict__ bias,
                         _Float16* __restrict__ actOut,
                         float* __restrict__ featOut,
                         int H, int W, int tilesX) {
  __shared__ __align__(16) _Float16 patch[CIN * 36];   // [cin][6][6] halo tile
  __shared__ __align__(16) _Float16 B_lds[16 * KPAD];  // [pixel n][k] im2col
  __shared__ float convout[64 * 16];                   // [ch][pixel] pre-pool

  const int img   = blockIdx.y;
  const int tileX = blockIdx.x % tilesX;
  const int tileY = blockIdx.x / tilesX;
  const int tid   = threadIdx.x;
  const int HW    = H * W;

  // ---- stage input halo patch (6x6 x CIN) into LDS, zero-padded ----
  const float*    imgF = nullptr;
  const _Float16* imgH = nullptr;
  if (LAYER1) {
    imgF = (img < 25)  ? (shot    + (size_t)img         * 3 * 84 * 84)
         : (img < 100) ? (query   + (size_t)(img - 25)  * 3 * 84 * 84)
                       : (unlabel + (size_t)(img - 100) * 3 * 84 * 84);
  } else {
    imgH = actIn + (size_t)img * CIN * HW;
  }
  for (int idx = tid; idx < CIN * 36; idx += 128) {
    int cin = idx / 36, rem = idx - cin * 36;
    int yy = rem / 6, xx = rem - yy * 6;
    int gy = tileY * 4 - 1 + yy, gx = tileX * 4 - 1 + xx;
    float val = 0.0f;
    if (gy >= 0 && gy < H && gx >= 0 && gx < W) {
      int o = cin * HW + gy * W + gx;
      val = LAYER1 ? imgF[o] : (float)imgH[o];
    }
    patch[idx] = (_Float16)val;
  }
  __syncthreads();

  // ---- build im2col B_lds[n][k]: decode k->(cin,ky,kx) exactly once ----
#pragma unroll
  for (int n = 0; n < 16; ++n) {
    const int py = n >> 2, px = n & 3;
    for (int k = tid; k < KPAD; k += 128) {
      const int cin = k / 9;
      const int r   = k - cin * 9;
      _Float16 v = (_Float16)0.0f;
      if (cin < CIN) v = patch[cin * 36 + (py + r / 3) * 6 + (px + r % 3)];
      B_lds[n * KPAD + k] = v;
    }
  }
  __syncthreads();

  // ---- K-loop: M=16 channels x N=16 pixels, K chunks of 32 ----
  const int wave = tid >> 5;
  const int lane = tid & 31;
  const int hl   = lane >> 4;  // lane half
  const int q    = lane & 15;  // M row (A) / pixel column (B)

  v8f acc = {0.f, 0.f, 0.f, 0.f, 0.f, 0.f, 0.f, 0.f};
  const _Float16* wrow = w16 + (size_t)(wave * 16 + q) * KPAD;
  const _Float16* bcol = &B_lds[q * KPAD];

#pragma unroll 4
  for (int kc = 0; kc < KPAD / 32; ++kc) {
    const v8h a0 = *(const v8h*)(wrow + kc * 32 + hl * 8);
    const v8h a1 = *(const v8h*)(wrow + kc * 32 + 16 + hl * 8);
    const v16h a = __builtin_shufflevector(a0, a1, 0, 1, 2, 3, 4, 5, 6, 7,
                                           8, 9, 10, 11, 12, 13, 14, 15);
    const v8h b0 = *(const v8h*)(bcol + kc * 32 + hl * 8);
    const v8h b1 = *(const v8h*)(bcol + kc * 32 + 16 + hl * 8);
    const v16h b = __builtin_shufflevector(b0, b1, 0, 1, 2, 3, 4, 5, 6, 7,
                                           8, 9, 10, 11, 12, 13, 14, 15);
    acc = __builtin_amdgcn_wmma_f32_16x16x32_f16(false, a, false, b,
                                                 (short)0, acc, false, false);
  }

  // ---- + bias, ReLU, spill conv tile to LDS ----
#pragma unroll
  for (int r = 0; r < 8; ++r) {
    const int ch = wave * 16 + hl * 8 + r;
    float v = acc[r] + bias[ch];
    convout[ch * 16 + q] = v > 0.f ? v : 0.f;
  }
  __syncthreads();

  // ---- 2x2 maxpool -> f16 activations (+ optional f32 features) ----
  const int pH = H >> 1, pW = W >> 1;
  for (int o = tid; o < 256; o += 128) {
    const int ch = o >> 2, pp = o & 3;
    const int py = pp >> 1, px = pp & 1;
    const float* c0 = &convout[ch * 16 + (py * 2) * 4 + px * 2];
    const float m = fmaxf(fmaxf(c0[0], c0[1]), fmaxf(c0[4], c0[5]));
    const int gpy = tileY * 2 + py, gpx = tileX * 2 + px;
    if (gpy < pH && gpx < pW) {
      actOut[(((size_t)img * 64 + ch) * pH + gpy) * pW + gpx] = (_Float16)m;
      if (featOut)
        featOut[(size_t)img * 1600 + ch * 25 + gpy * 5 + gpx] = m;
    }
  }
}

// ---------------------------------------------------------------------------
// proto_shot[w][d] = mean over 5 shots of feats[(s*5+w)][d]
// ---------------------------------------------------------------------------
__global__ void k_proto_shot(const float* __restrict__ feats, float* __restrict__ ps) {
  int idx = blockIdx.x * blockDim.x + threadIdx.x;
  if (idx >= 5 * 1600) return;
  int w = idx / 1600, d = idx - w * 1600;
  float s = 0.f;
#pragma unroll
  for (int sh = 0; sh < 5; ++sh) s += feats[(size_t)(sh * 5 + w) * 1600 + d];
  ps[idx] = s * 0.2f;
}

// ---------------------------------------------------------------------------
// out[i][j] = -sum_d (feats[rowBase+i][d] - protos[j][d])^2   (DIS_TAU = 1)
// ---------------------------------------------------------------------------
__global__ __launch_bounds__(256)
void k_pair_dist(const float* __restrict__ feats, const float* __restrict__ protos,
                 float* __restrict__ out, int rowBase) {
  __shared__ float s[256];
  const int i = blockIdx.x, t = threadIdx.x;
  const float* f = feats + (size_t)(rowBase + i) * 1600;
  float acc[5] = {0.f, 0.f, 0.f, 0.f, 0.f};
  for (int d = t; d < 1600; d += 256) {
    const float fv = f[d];
#pragma unroll
    for (int j = 0; j < 5; ++j) {
      const float df = fv - protos[j * 1600 + d];
      acc[j] += df * df;
    }
  }
#pragma unroll
  for (int j = 0; j < 5; ++j) {
    s[t] = acc[j];
    __syncthreads();
    for (int o = 128; o > 0; o >>= 1) {
      if (t < o) s[t] += s[t + o];
      __syncthreads();
    }
    if (t == 0) out[i * 5 + j] = -s[0];
    __syncthreads();
  }
}

// ---------------------------------------------------------------------------
// Column statistics over 256 rows: sd = (-dis)/colmean(-dis); stats row j =
// [min, max, m2, skew, kurt] of sd column j. One block per column.
// ---------------------------------------------------------------------------
__global__ __launch_bounds__(256)
void k_stats(const float* __restrict__ dis, float* __restrict__ sd,
             float* __restrict__ stats) {
  __shared__ float s[256];
  const int j = blockIdx.x, t = threadIdx.x;
  const float v = -dis[t * 5 + j];

  s[t] = v; __syncthreads();
  for (int o = 128; o > 0; o >>= 1) { if (t < o) s[t] += s[t + o]; __syncthreads(); }
  const float meanv = s[0] * (1.0f / 256.0f); __syncthreads();

  const float sdv = v / meanv;
  sd[t * 5 + j] = sdv;

  s[t] = sdv; __syncthreads();
  for (int o = 128; o > 0; o >>= 1) { if (t < o) s[t] += s[t + o]; __syncthreads(); }
  const float mu = s[0] * (1.0f / 256.0f); __syncthreads();

  const float c = sdv - mu;
  s[t] = c * c; __syncthreads();
  for (int o = 128; o > 0; o >>= 1) { if (t < o) s[t] += s[t + o]; __syncthreads(); }
  const float m2 = s[0] * (1.0f / 256.0f); __syncthreads();

  s[t] = c * c * c; __syncthreads();
  for (int o = 128; o > 0; o >>= 1) { if (t < o) s[t] += s[t + o]; __syncthreads(); }
  const float m3 = s[0] * (1.0f / 256.0f); __syncthreads();

  s[t] = c * c * c * c; __syncthreads();
  for (int o = 128; o > 0; o >>= 1) { if (t < o) s[t] += s[t + o]; __syncthreads(); }
  const float m4 = s[0] * (1.0f / 256.0f); __syncthreads();

  s[t] = sdv; __syncthreads();
  for (int o = 128; o > 0; o >>= 1) { if (t < o) s[t] = fminf(s[t], s[t + o]); __syncthreads(); }
  const float mn = s[0]; __syncthreads();

  s[t] = sdv; __syncthreads();
  for (int o = 128; o > 0; o >>= 1) { if (t < o) s[t] = fmaxf(s[t], s[t + o]); __syncthreads(); }
  const float mx = s[0]; __syncthreads();

  if (t == 0) {
    stats[j * 5 + 0] = mn;
    stats[j * 5 + 1] = mx;
    stats[j * 5 + 2] = m2;
    stats[j * 5 + 3] = m3 / (m2 * sqrtf(m2));
    stats[j * 5 + 4] = m4 / (m2 * m2) - 3.0f;
  }
}

// ---------------------------------------------------------------------------
// detector MLP: 5 rows, (5 -> 3 relu -> 2). bg[0..4]=beta, bg[5..9]=gamma
// ---------------------------------------------------------------------------
__global__ void k_det(const float* __restrict__ stats, const float* __restrict__ w1,
                      const float* __restrict__ b1, const float* __restrict__ w2,
                      const float* __restrict__ b2, float* __restrict__ bg) {
  const int j = threadIdx.x;
  if (j >= 5) return;
  float h[3];
#pragma unroll
  for (int m = 0; m < 3; ++m) {
    float s = b1[m];
#pragma unroll
    for (int k = 0; k < 5; ++k) s += stats[j * 5 + k] * w1[k * 3 + m];
    h[m] = s > 0.f ? s : 0.f;
  }
  float o0 = b2[0], o1 = b2[1];
#pragma unroll
  for (int m = 0; m < 3; ++m) { o0 += h[m] * w2[m * 2 + 0]; o1 += h[m] * w2[m * 2 + 1]; }
  bg[j] = o0;       // beta
  bg[5 + j] = o1;   // gamma
}

// ---------------------------------------------------------------------------
// z_hat[i][j] = softmax(dis[i,:])[j] * sigmoid(-gamma_j*(sd[i][j]-beta_j))
// ---------------------------------------------------------------------------
__global__ __launch_bounds__(256)
void k_zhat(const float* __restrict__ dis, const float* __restrict__ sd,
            const float* __restrict__ bg, float* __restrict__ zhat) {
  const int i = blockIdx.x * blockDim.x + threadIdx.x;
  if (i >= 256) return;
  float d[5], mx = -1e30f;
#pragma unroll
  for (int j = 0; j < 5; ++j) { d[j] = dis[i * 5 + j]; mx = fmaxf(mx, d[j]); }
  float e[5], se = 0.f;
#pragma unroll
  for (int j = 0; j < 5; ++j) { e[j] = __expf(d[j] - mx); se += e[j]; }
#pragma unroll
  for (int j = 0; j < 5; ++j) {
    const float sm   = e[j] / se;
    const float madj = 1.0f / (1.0f + __expf(bg[5 + j] * (sd[i * 5 + j] - bg[j])));
    zhat[i * 5 + j] = sm * madj;
  }
}

// ---------------------------------------------------------------------------
// proto[j][d] = (sum_s x_shot[s*5+j][d] + sum_i zhat[i][j]*x_pool[i][d])
//               / (5 + sum_i zhat[i][j])
// ---------------------------------------------------------------------------
__global__ void k_proto(const float* __restrict__ feats, const float* __restrict__ zhat,
                        float* __restrict__ proto) {
  const int idx = blockIdx.x * blockDim.x + threadIdx.x;
  if (idx >= 5 * 1600) return;
  const int j = idx / 1600, d = idx - j * 1600;
  float num = 0.f, den = 5.0f;
#pragma unroll
  for (int sh = 0; sh < 5; ++sh) num += feats[(size_t)(sh * 5 + j) * 1600 + d];
  for (int i = 0; i < 256; ++i) {
    const float z = zhat[i * 5 + j];
    num += z * feats[(size_t)(100 + i) * 1600 + d];
    den += z;
  }
  proto[idx] = num / den;
}

// ---------------------------------------------------------------------------
extern "C" void kernel_launch(void* const* d_in, const int* in_sizes, int n_in,
                              void* d_out, int out_size, void* d_ws, size_t ws_size,
                              hipStream_t stream) {
  // ---- identify flattened inputs by element count (pytree-order agnostic) ----
  int i_shot = -1, i_query = -1, i_unl = -1;
  int i_w[4] = {-1, -1, -1, -1};
  int i64[8]; int n64 = 0, n36 = 0;
  int i_dw1 = -1, i_db1 = -1, i_dw2 = -1, i_db2 = -1;
  for (int i = 0; i < n_in; ++i) {
    const long s = in_sizes[i];
    if      (s == 25L  * 3 * 84 * 84) i_shot  = i;
    else if (s == 75L  * 3 * 84 * 84) i_query = i;
    else if (s == 256L * 3 * 84 * 84) i_unl   = i;
    else if (s == 1728)               i_w[0]  = i;
    else if (s == 36864 && n36 < 3)   i_w[1 + n36++] = i;
    else if (s == 64 && n64 < 8)      i64[n64++] = i;
    else if (s == 15)                 i_dw1 = i;
    else if (s == 3)                  i_db1 = i;
    else if (s == 6)                  i_dw2 = i;
    else if (s == 2)                  i_db2 = i;
  }
  if (i_shot < 0 || i_query < 0 || i_unl < 0 || i_w[0] < 0 || n36 < 3 || n64 < 8 ||
      i_dw1 < 0 || i_db1 < 0 || i_dw2 < 0 || i_db2 < 0)
    return;

  // sorted-key flattening puts bn_bias,bn_scale before conv_w; insertion order
  // puts conv_w first then bn_scale,bn_bias.
  const bool biasFirst = (i64[0] < i_w[0]);
  const float *scaleP[4], *biasP[4], *convP[4];
  for (int l = 0; l < 4; ++l) {
    biasP[l]  = (const float*)d_in[biasFirst ? i64[l]     : i64[4 + l]];
    scaleP[l] = (const float*)d_in[biasFirst ? i64[4 + l] : i64[l]];
    convP[l]  = (const float*)d_in[i_w[l]];
  }
  const float* shot  = (const float*)d_in[i_shot];
  const float* query = (const float*)d_in[i_query];
  const float* unl   = (const float*)d_in[i_unl];
  const float* dw1   = (const float*)d_in[i_dw1];
  const float* db1   = (const float*)d_in[i_db1];
  const float* dw2   = (const float*)d_in[i_dw2];
  const float* db2   = (const float*)d_in[i_db2];

  // ---- carve workspace ----
  char* ws = (char*)d_ws;
  size_t off = 0;
  auto alloc = [&](size_t bytes) -> void* {
    off = (off + 255) & ~(size_t)255;
    void* p = ws + off;
    off += bytes;
    return p;
  };
  const int NIMG = 356;
  _Float16* w16[4];
  w16[0] = (_Float16*)alloc((size_t)64 * 32  * 2);
  for (int l = 1; l < 4; ++l) w16[l] = (_Float16*)alloc((size_t)64 * 576 * 2);
  _Float16* act1 = (_Float16*)alloc((size_t)NIMG * 64 * 42 * 42 * 2);
  _Float16* act2 = (_Float16*)alloc((size_t)NIMG * 64 * 21 * 21 * 2);
  _Float16* act3 = (_Float16*)alloc((size_t)NIMG * 64 * 10 * 10 * 2);
  _Float16* act4 = (_Float16*)alloc((size_t)NIMG * 64 * 5 * 5 * 2);
  float* feats   = (float*)alloc((size_t)NIMG * 1600 * 4);
  float* pshot   = (float*)alloc(5 * 1600 * 4);
  float* dis     = (float*)alloc(256 * 5 * 4);
  float* sd      = (float*)alloc(256 * 5 * 4);
  float* stats   = (float*)alloc(5 * 5 * 4);
  float* bg      = (float*)alloc(10 * 4);
  float* zhat    = (float*)alloc(256 * 5 * 4);
  float* proto   = (float*)alloc(5 * 1600 * 4);
  (void)ws_size;

  // ---- weights -> f16 with BN scale folded, K-padded ----
  k_cvt_weights<<<(64 * 32  + 255) / 256, 256, 0, stream>>>(convP[0], scaleP[0], w16[0], 27,  32);
  for (int l = 1; l < 4; ++l)
    k_cvt_weights<<<(64 * 576 + 255) / 256, 256, 0, stream>>>(convP[l], scaleP[l], w16[l], 576, 576);

  // ---- encoder: 4x (conv + bn + relu + pool), WMMA implicit GEMM ----
  k_conv_bn_relu_pool<3, 32, true><<<dim3(21 * 21, NIMG), 128, 0, stream>>>(
      shot, query, unl, (const _Float16*)nullptr, w16[0], biasP[0],
      act1, (float*)nullptr, 84, 84, 21);
  k_conv_bn_relu_pool<64, 576, false><<<dim3(11 * 11, NIMG), 128, 0, stream>>>(
      nullptr, nullptr, nullptr, act1, w16[1], biasP[1],
      act2, (float*)nullptr, 42, 42, 11);
  k_conv_bn_relu_pool<64, 576, false><<<dim3(6 * 6, NIMG), 128, 0, stream>>>(
      nullptr, nullptr, nullptr, act2, w16[2], biasP[2],
      act3, (float*)nullptr, 21, 21, 6);
  k_conv_bn_relu_pool<64, 576, false><<<dim3(3 * 3, NIMG), 128, 0, stream>>>(
      nullptr, nullptr, nullptr, act3, w16[3], biasP[3],
      act4, feats, 10, 10, 3);

  // ---- tail ----
  k_proto_shot<<<(5 * 1600 + 255) / 256, 256, 0, stream>>>(feats, pshot);
  k_pair_dist<<<256, 256, 0, stream>>>(feats, pshot, dis, 100);      // pool vs proto_shot
  k_stats<<<5, 256, 0, stream>>>(dis, sd, stats);
  k_det<<<1, 32, 0, stream>>>(stats, dw1, db1, dw2, db2, bg);
  k_zhat<<<1, 256, 0, stream>>>(dis, sd, bg, zhat);
  k_proto<<<(5 * 1600 + 255) / 256, 256, 0, stream>>>(feats, zhat, proto);
  k_pair_dist<<<75, 256, 0, stream>>>(feats, proto, (float*)d_out, 25); // query vs proto
}
